// MultiHeadSelfAttention_57234734186503
// MI455X (gfx1250) — compile-verified, matrix operand order checked by reading, not run on hardware
//
#include <hip/hip_runtime.h>
#include <stdint.h>

#define DEV __device__ __forceinline__

typedef __bf16 bf16_t;
typedef bf16_t v16bf __attribute__((ext_vector_type(16)));
typedef float  v8f   __attribute__((ext_vector_type(8)));

static constexpr int S  = 4096;
static constexpr int D  = 256;
static constexpr int H  = 8;
static constexpr int Dh = 32;
static constexpr int NTOK = 2 * S;           // B*S = 8192 tokens
static constexpr float INV_SQRT_DH = 0.17677669529663687f;   // 1/sqrt(32)
static constexpr float LOG2E       = 1.4426950408889634f;
static constexpr float SA          = INV_SQRT_DH * LOG2E;    // score scale (exp2 domain)
static constexpr float KBIAS_SCALE = -1.0e9f * INV_SQRT_DH * LOG2E;
static constexpr float LN_EPS      = 1e-6f;

union ABTile { v16bf v; uint4 q[2]; };
union Pack8  { bf16_t b[8]; uint4 q; };

// ------------------------------------------------------- CDNA5 async helpers
// Async copy 16B global -> LDS (GLOBAL_LOAD_ASYNC_TO_LDS_B128, ASYNCcnt).
// LDS operand is the low 32 bits of the flat address (LDS aperture is in the
// high word per ISA 10.2 aperture mapping).
DEV void async_copy16(const bf16_t* g, bf16_t* l) {
    uint32_t loff = (uint32_t)(uintptr_t)l;
    uint64_t ga   = (uint64_t)(uintptr_t)g;
    asm volatile("global_load_async_to_lds_b128 %0, %1, off"
                 :: "v"(loff), "v"(ga) : "memory");
}

template <int N>
DEV void wait_async() {
    asm volatile("s_wait_asynccnt %0" :: "n"(N) : "memory");
}

// Load one 16x32 bf16 WMMA operand tile (A or B side; both use the same lane
// pattern): row = base row + (lane&15); half (lane>=16) takes k-elements
// {8..15, 24..31}, half 0 takes {0..7, 16..23}. Two 16B loads per lane.
DEV v16bf load_tile(const bf16_t* __restrict__ base, int row, int rowStride,
                    int kBase, int lane) {
    const int r  = lane & 15;
    const int hl = lane >> 4;
    const bf16_t* p = base + (size_t)(row + r) * rowStride + kBase + hl * 8;
    ABTile t;
    t.q[0] = *(const uint4*)(p);
    t.q[1] = *(const uint4*)(p + 16);
    return t.v;
}

// Same tile fetch but from an LDS tile with row stride 32 elements.
DEV v16bf load_tile_lds(const bf16_t* base, int row, int lane) {
    const int r  = lane & 15;
    const int hl = lane >> 4;
    const bf16_t* p = base + (row + r) * 32 + hl * 8;
    ABTile t;
    t.q[0] = *(const uint4*)(p);
    t.q[1] = *(const uint4*)(p + 16);
    return t.v;
}

// ---------------------------------------------------------------- prep kernels
__global__ void __launch_bounds__(256) cvt_x_kernel(const float* __restrict__ x,
                                                    bf16_t* __restrict__ xb) {
    int i = blockIdx.x * 256 + threadIdx.x;        // 8192*256 elements
    xb[i] = (bf16_t)x[i];
}

__global__ void __launch_bounds__(256) wt_kernel(const float* __restrict__ Wq,
                                                 const float* __restrict__ Wk,
                                                 const float* __restrict__ Wv,
                                                 bf16_t* __restrict__ wt3) {
    int i    = blockIdx.x * 256 + threadIdx.x;     // 3*256*256 elements
    int proj = i >> 16;
    int rem  = i & 65535;
    int n    = rem >> 8;
    int k    = rem & 255;
    const float* W = (proj == 0) ? Wq : ((proj == 1) ? Wk : Wv);
    wt3[i] = (bf16_t)W[k * 256 + n];               // wt3[proj][n][k] = W[k][n]
}

__global__ void __launch_bounds__(256) kbias_kernel(const float* __restrict__ pm,
                                                    float* __restrict__ kb2) {
    int i = blockIdx.x * 256 + threadIdx.x;        // 8192 elements
    kb2[i] = pm[i] * KBIAS_SCALE;
}

// -------------------------------------------------------------- QKV projection
// One wave computes a 16x16 output tile of one projection (M=8192, N=256,
// K=256), 8 x v_wmma_f32_16x16x32_bf16 per tile, bias preloaded into C.
__global__ void __launch_bounds__(256) qkv_kernel(
    const bf16_t* __restrict__ xb,    // [8192][256]
    const bf16_t* __restrict__ wt3,   // [3][256 n][256 k]  (W transposed)
    const float* __restrict__ bq, const float* __restrict__ bk,
    const float* __restrict__ bv,
    bf16_t* __restrict__ Qo,          // [B,H,S,32]
    bf16_t* __restrict__ Ko,          // [B,H,S,32]
    bf16_t* __restrict__ Vto) {       // [B,H,32,S]  (V transposed)
    const int lane = threadIdx.x & 31;
    const int wave = threadIdx.x >> 5;
    const int w    = blockIdx.x * 8 + wave;        // 0..24575
    const int proj = w >> 13;
    const int rem  = w & 8191;
    const int m0   = (rem >> 4) * 16;              // token tile base
    const int n0   = (rem & 15) * 16;              // out-dim tile base
    const int r    = lane & 15;
    const int hl   = lane >> 4;

    const bf16_t* wtp  = wt3 + (size_t)proj * 65536;
    const float*  bias = (proj == 0) ? bq : ((proj == 1) ? bk : bv);
    const float   bval = bias[n0 + r];

    v8f c;
#pragma unroll
    for (int j = 0; j < 8; ++j) c[j] = bval;

#pragma unroll
    for (int kc = 0; kc < 8; ++kc) {
        v16bf a  = load_tile(xb,  m0, 256, kc * 32, lane);
        v16bf bt = load_tile(wtp, n0, 256, kc * 32, lane);
        c = __builtin_amdgcn_wmma_f32_16x16x32_bf16(false, a, false, bt,
                                                    (short)0, c, false, false);
    }

    // epilogue: lane holds column n = n0+r; rows m = m0 + j + 8*hl
    const int n     = n0 + r;
    const int hh    = n >> 5;
    const int d     = n & 31;
    const int bb    = m0 >> 12;                    // batch (tile never straddles)
    const int sbase = m0 & 4095;
    const size_t bh = (size_t)(bb * 8 + hh);

    if (proj == 2) {
        // V transposed: [bh][d][s] -> 8 consecutive s per lane: one 16B store
        Pack8 pk;
#pragma unroll
        for (int j = 0; j < 8; ++j) pk.b[j] = (bf16_t)c[j];
        bf16_t* dst = Vto + (bh * 32 + d) * S + sbase + hl * 8;
        *(uint4*)dst = pk.q;
    } else {
        bf16_t* base = ((proj == 0) ? Qo : Ko) + (bh * S + sbase) * 32 + d;
#pragma unroll
        for (int j = 0; j < 8; ++j)
            base[(size_t)(j + 8 * hl) * 32] = (bf16_t)c[j];
    }
}

// ------------------------------------------------------------ flash attention
// 8 waves per workgroup share one (b,h); each wave owns a 16-query tile.
// K/V chunks (32 keys) are staged global->LDS with async-to-LDS copies, double
// buffered: each of the 256 threads copies one 16B segment, i.e. exactly one
// async instruction per wave per chunk, so per-wave ASYNCcnt is 0/1.
// Scores are computed transposed (S^T = K_tile x Q^T) so softmax reductions
// are per-lane and the exp() results drop directly into the B-operand layout
// of the second WMMA (O^T = V^T x P^T). 4 WMMA per 32-key chunk.
__global__ void __launch_bounds__(256) attn_kernel(
    const bf16_t* __restrict__ Q,   // [B,H,S,32]
    const bf16_t* __restrict__ K,   // [B,H,S,32]
    const bf16_t* __restrict__ Vt,  // [B,H,32,S]
    const float* __restrict__ kb2,  // [B,S] additive key bias (exp2 domain)
    float* __restrict__ O) {        // [B,S,256]
    __shared__ bf16_t ldsK[2][32 * 32];   // [buf][key_local][32]   2KB each
    __shared__ bf16_t ldsV[2][32 * 32];   // [buf][d][key_local]    2KB each

    const int tid  = threadIdx.x;
    const int lane = tid & 31;
    const int wave = tid >> 5;
    const int bh   = blockIdx.y;                   // 0..15
    const int q0   = (blockIdx.x * 8 + wave) * 16; // query tile base
    const int b    = bh >> 3;
    const int h    = bh & 7;
    const int r    = lane & 15;
    const int hl   = lane >> 4;

    const bf16_t* Qb = Q  + (size_t)bh * S * Dh;
    const bf16_t* Kb = K  + (size_t)bh * S * Dh;
    const bf16_t* Vb = Vt + (size_t)bh * Dh * S;
    const float*  kb = kb2 + (size_t)b * S;

    // cooperative stage of one 32-key chunk: threads 0..127 -> K (32 rows x
    // 64B), threads 128..255 -> V^T (32 rows x 64B); 16B per thread.
    const int sidx = tid & 127;
    const int srow = sidx >> 2;
    const int sseg = (sidx & 3) * 8;               // element offset within row
    auto stage = [&](int kbase, int buf) {
        if (tid < 128)
            async_copy16(Kb + (size_t)(kbase + srow) * Dh + sseg,
                         &ldsK[buf][srow * 32 + sseg]);
        else
            async_copy16(Vb + (size_t)srow * S + kbase + sseg,
                         &ldsV[buf][srow * 32 + sseg]);
    };

    const v16bf bqop = load_tile(Qb, q0, Dh, 0, lane);   // Q^T operand, reused

    v8f o0, o1;
#pragma unroll
    for (int j = 0; j < 8; ++j) { o0[j] = 0.f; o1[j] = 0.f; }
    float m_run = -3.0e38f;
    float l_run = 0.f;

    const int nChunks = S / 32;
    stage(0, 0);
    for (int kc = 0; kc < nChunks; ++kc) {
        const int buf = kc & 1;
        if (kc + 1 < nChunks) {
            stage((kc + 1) * 32, buf ^ 1);
            wait_async<1>();                       // chunk kc landed, kc+1 in flight
        } else {
            wait_async<0>();
        }
        __syncthreads();                           // all waves' staging visible

        v16bf ak0 = load_tile_lds(ldsK[buf], 0,  lane);  // keys +0..15
        v16bf ak1 = load_tile_lds(ldsK[buf], 16, lane);  // keys +16..31
        v16bf av0 = load_tile_lds(ldsV[buf], 0,  lane);  // V^T d 0..15
        v16bf av1 = load_tile_lds(ldsV[buf], 16, lane);  // V^T d 16..31

        v8f z;
#pragma unroll
        for (int j = 0; j < 8; ++j) z[j] = 0.f;
        v8f st0 = __builtin_amdgcn_wmma_f32_16x16x32_bf16(false, ak0, false, bqop,
                                                          (short)0, z, false, false);
        v8f st1 = __builtin_amdgcn_wmma_f32_16x16x32_bf16(false, ak1, false, bqop,
                                                          (short)0, z, false, false);

        // st*[j] = dot(key = kbase + j + 8*hl (+16), query = q0+r)
        const int kbase = kc * 32;
        const float* kbp = kb + kbase;
        float w0[8], w1[8];
        float tmax = -3.0e38f;
#pragma unroll
        for (int j = 0; j < 8; ++j) {
            float b0 = hl ? kbp[j + 8]  : kbp[j];        // uniform scalar loads
            float b1 = hl ? kbp[j + 24] : kbp[j + 16];
            w0[j] = fmaf(st0[j], SA, b0);
            w1[j] = fmaf(st1[j], SA, b1);
            tmax  = fmaxf(tmax, fmaxf(w0[j], w1[j]));
        }
        tmax = fmaxf(tmax, __shfl_xor(tmax, 16));        // combine key halves
        const float m_new = fmaxf(m_run, tmax);
        const float alpha = __builtin_amdgcn_exp2f(m_run - m_new);
        m_run = m_new;

        Pack8 p0, p1;
        float psum = 0.f;
#pragma unroll
        for (int j = 0; j < 8; ++j) {
            float e0 = __builtin_amdgcn_exp2f(w0[j] - m_new);
            float e1 = __builtin_amdgcn_exp2f(w1[j] - m_new);
            psum += e0 + e1;
            p0.b[j] = (bf16_t)e0;
            p1.b[j] = (bf16_t)e1;
        }
        l_run = l_run * alpha + psum;
#pragma unroll
        for (int j = 0; j < 8; ++j) { o0[j] *= alpha; o1[j] *= alpha; }

        // P^T already matches the B-operand layout: no shuffles needed.
        ABTile pb;
        pb.q[0] = p0.q;
        pb.q[1] = p1.q;
        o0 = __builtin_amdgcn_wmma_f32_16x16x32_bf16(false, av0, false, pb.v,
                                                     (short)0, o0, false, false);
        o1 = __builtin_amdgcn_wmma_f32_16x16x32_bf16(false, av1, false, pb.v,
                                                     (short)0, o1, false, false);

        __syncthreads();                           // buffer reuse next iteration
    }

    l_run += __shfl_xor(l_run, 16);
    const float inv_l = 1.0f / l_run;

    // O^T layout: lane holds query q0+r; d = j + 8*hl (o0) / 16 + j + 8*hl (o1)
    float* orow = O + (size_t)b * S * D + (size_t)(q0 + r) * D + h * Dh + hl * 8;
    float4 f0 = make_float4(o0[0] * inv_l, o0[1] * inv_l, o0[2] * inv_l, o0[3] * inv_l);
    float4 f1 = make_float4(o0[4] * inv_l, o0[5] * inv_l, o0[6] * inv_l, o0[7] * inv_l);
    float4 f2 = make_float4(o1[0] * inv_l, o1[1] * inv_l, o1[2] * inv_l, o1[3] * inv_l);
    float4 f3 = make_float4(o1[4] * inv_l, o1[5] * inv_l, o1[6] * inv_l, o1[7] * inv_l);
    *(float4*)(orow)      = f0;
    *(float4*)(orow + 4)  = f1;
    *(float4*)(orow + 16) = f2;
    *(float4*)(orow + 20) = f3;
}

// --------------------------------------------------------- residual + LayerNorm
__global__ void __launch_bounds__(256) ln_kernel(const float* __restrict__ O,
                                                 const float* __restrict__ x,
                                                 const float* __restrict__ gamma,
                                                 const float* __restrict__ beta,
                                                 float* __restrict__ out) {
    const int lane = threadIdx.x & 31;
    const int wave = threadIdx.x >> 5;
    const int t    = blockIdx.x * 8 + wave;        // token

    const float4* o4 = (const float4*)(O + (size_t)t * D) + lane * 2;
    const float4* x4 = (const float4*)(x + (size_t)t * D) + lane * 2;
    float4 a0 = o4[0], a1 = o4[1], c0 = x4[0], c1 = x4[1];
    float y[8] = {a0.x + c0.x, a0.y + c0.y, a0.z + c0.z, a0.w + c0.w,
                  a1.x + c1.x, a1.y + c1.y, a1.z + c1.z, a1.w + c1.w};

    float s = 0.f;
#pragma unroll
    for (int j = 0; j < 8; ++j) s += y[j];
#pragma unroll
    for (int off = 16; off; off >>= 1) s += __shfl_xor(s, off);
    const float mu = s * (1.0f / 256.0f);

    float ss = 0.f;
#pragma unroll
    for (int j = 0; j < 8; ++j) { float dd = y[j] - mu; ss += dd * dd; }
#pragma unroll
    for (int off = 16; off; off >>= 1) ss += __shfl_xor(ss, off);
    const float rs = rsqrtf(ss * (1.0f / 256.0f) + LN_EPS);

    const float4* g4 = (const float4*)gamma + lane * 2;
    const float4* b4 = (const float4*)beta + lane * 2;
    float4 g0 = g4[0], g1 = g4[1], be0 = b4[0], be1 = b4[1];

    float4 r0 = make_float4(g0.x * (y[0] - mu) * rs + be0.x,
                            g0.y * (y[1] - mu) * rs + be0.y,
                            g0.z * (y[2] - mu) * rs + be0.z,
                            g0.w * (y[3] - mu) * rs + be0.w);
    float4 r1 = make_float4(g1.x * (y[4] - mu) * rs + be1.x,
                            g1.y * (y[5] - mu) * rs + be1.y,
                            g1.z * (y[6] - mu) * rs + be1.z,
                            g1.w * (y[7] - mu) * rs + be1.w);
    float4* out4 = (float4*)(out + (size_t)t * D) + lane * 2;
    out4[0] = r0;
    out4[1] = r1;
}

// ------------------------------------------------------------------- launcher
extern "C" void kernel_launch(void* const* d_in, const int* in_sizes, int n_in,
                              void* d_out, int out_size, void* d_ws, size_t ws_size,
                              hipStream_t stream) {
    (void)in_sizes; (void)n_in; (void)out_size; (void)ws_size;
    const float* x     = (const float*)d_in[0];
    const float* pm    = (const float*)d_in[1];
    const float* Wq    = (const float*)d_in[2];
    const float* bq    = (const float*)d_in[3];
    const float* Wk    = (const float*)d_in[4];
    const float* bk    = (const float*)d_in[5];
    const float* Wv    = (const float*)d_in[6];
    const float* bv    = (const float*)d_in[7];
    const float* gamma = (const float*)d_in[8];
    const float* beta  = (const float*)d_in[9];
    float* out = (float*)d_out;

    char*  ws  = (char*)d_ws;
    size_t off = 0;
    auto alloc = [&](size_t bytes) {
        void* p = ws + off;
        off += (bytes + 255) & ~(size_t)255;
        return p;
    };
    bf16_t* xb  = (bf16_t*)alloc((size_t)NTOK * D * 2);       // 4 MB
    bf16_t* wt3 = (bf16_t*)alloc((size_t)3 * 256 * 256 * 2);  // 384 KB
    float*  kb2 = (float*)alloc((size_t)NTOK * 4);            // 32 KB
    bf16_t* Qb  = (bf16_t*)alloc((size_t)16 * S * Dh * 2);    // 4 MB
    bf16_t* Kb  = (bf16_t*)alloc((size_t)16 * S * Dh * 2);    // 4 MB
    bf16_t* Vt  = (bf16_t*)alloc((size_t)16 * Dh * S * 2);    // 4 MB
    float*  Ob  = (float*)alloc((size_t)NTOK * D * 4);        // 8 MB

    cvt_x_kernel<<<NTOK * D / 256, 256, 0, stream>>>(x, xb);
    wt_kernel<<<3 * 256 * 256 / 256, 256, 0, stream>>>(Wq, Wk, Wv, wt3);
    kbias_kernel<<<NTOK / 256, 256, 0, stream>>>(pm, kb2);
    qkv_kernel<<<3072, 256, 0, stream>>>(xb, wt3, bq, bk, bv, Qb, Kb, Vt);
    attn_kernel<<<dim3(32, 16), 256, 0, stream>>>(Qb, Kb, Vt, kb2, Ob);
    ln_kernel<<<NTOK / 8, 256, 0, stream>>>(Ob, x, gamma, beta, out);
}
